// SDT_9070970929349
// MI455X (gfx1250) — matrix operand order repaired and checked by reading, not compiled, not run on hardware
//
#include <hip/hip_runtime.h>

// Soft Decision Tree forward, fused for MI455X (gfx1250, wave32, WMMA).
//   Kernel 0: zero 256-float leaf-sum scratch (in d_ws)
//   Kernel P: pre-swizzle + bf16-convert W_inner into per-chunk B-fragment
//             LDS images (in d_ws) -- hoists all W staging ALU out of hot loop
//   Kernel 1: bf16 WMMA GEMM (Xa @ W_inner^T) + sigmoid + tree walk + y_pred
//             + per-leaf batch-sum reduction (LDS atomics -> global atomics)
//   Kernel 2: heap-sum rebuild + log penalty -> d_out[B*10]
//
// v4: double-buffered pipeline. Per chunk: ONE barrier; chunk kt+1's W copy
// (global_load_async_to_lds_b128, ASYNCcnt) and X stage overlap the 8 WMMAs
// of chunk kt. Safety: each wave drains its own ASYNCcnt before the barrier,
// and the barrier at iteration kt orders every wave's compute(kt-1) before
// anyone re-stages that buffer at kt+1.

#define USE_ASYNC_COPY 1

#define B_TOTAL   131072
#define IN_DIM    784
#define KDIM      785          // IN_DIM + 1 (bias column)
#define NCHUNK    25           // ceil(785/32), K padded to 800 with zeros
#define INTERNAL  255
#define NCOLS     256          // padded node columns
#define LEAF      256
#define OUT_DIM   10
#define MTILE     64           // rows per workgroup = 4 row-blocks x 16
#define OUT_ROWS_PER_WAVE 8
#define LAMDA     1e-3f

typedef __attribute__((ext_vector_type(16))) __bf16  v16bf;
typedef __attribute__((ext_vector_type(2)))  __bf16  bf16x2;
typedef __attribute__((ext_vector_type(8)))  float   v8f;

// Dynamic LDS layout (bytes), double-buffered fragments:
//   wbuf[2] : bf16[16 tiles][32 lanes][16]  B fragments (async-copied)  (2 x 16 KB)
//   abuf[2] : bf16[4 rowblk][32 lanes][16]  A fragments (cvt bf16)      (2 x  4 KB)
//   p_lds   : float[4 rowblk][16 rows][256]                             (64 KB)
//   wleaf   : float[10*256]                                             (10 KB)
//   sleaf   : float[256]                                                ( 1 KB)
#define SM_WBUF   0
#define SM_ABUF   (2 * 16 * 1024)
#define SM_P      (SM_ABUF + 2 * 4096)
#define SM_WLEAF  (SM_P + 4 * 16 * NCOLS * 4)
#define SM_SLEAF  (SM_WLEAF + OUT_DIM * LEAF * 4)
#define SM_BYTES  (SM_SLEAF + LEAF * 4)

// d_ws layout: [0,1K) sleaf partials; [1K, 1K+400K) pre-swizzled W (25 x 16 KB)
#define WS_WSWZ_OFF  1024
#define WUNITS       (NCHUNK * 4096)   // u32 units total

__global__ void sdt_zero(float* __restrict__ S) {
    if (threadIdx.x < LEAF) S[threadIdx.x] = 0.0f;
}

// One-time: W_inner [255,785] f32 -> bf16, laid out as the exact 16 KB LDS
// image each K-chunk needs (B 32x16, this code's k order, same as A: lane(l<16)=col, reg r
// holds K pair (2r,2r+1); lanes>=16 hold K+16).
__global__ void sdt_prep_w(const float* __restrict__ Wi,
                           bf16x2* __restrict__ Wswz) {
    const int id = blockIdx.x * 256 + threadIdx.x;
    if (id >= WUNITS) return;
    const int c = id >> 12;            // chunk
    const int u = id & 4095;
    const int l = (u >> 3) & 31;
    const int r = u & 7;
    const int col = ((u >> 8) << 4) + (l & 15);
    const int k0  = c * 32 + 2 * r + ((l & 16) ? 16 : 0);
    const float f0 = (col < INTERNAL && k0     < KDIM) ? Wi[(size_t)col * KDIM + k0]     : 0.0f;
    const float f1 = (col < INTERNAL && k0 + 1 < KDIM) ? Wi[(size_t)col * KDIM + k0 + 1] : 0.0f;
    bf16x2 pk; pk.x = (__bf16)f0; pk.y = (__bf16)f1;
    Wswz[id] = pk;
}

__global__ void __launch_bounds__(256)
sdt_main(const float* __restrict__ X,
         const bf16x2* __restrict__ Wswz,  // pre-swizzled W chunks
         const float* __restrict__ Wl,     // [10, 256]
         float* __restrict__ out,          // [B,10] then penalty
         float* __restrict__ Sleaf_g) {    // [256] scratch
    extern __shared__ char smem[];
    __bf16* wbuf  = (__bf16*)(smem + SM_WBUF);
    __bf16* abuf  = (__bf16*)(smem + SM_ABUF);
    float*  p_lds = (float*) (smem + SM_P);
    float*  wleaf = (float*) (smem + SM_WLEAF);
    float*  sleaf = (float*) (smem + SM_SLEAF);

    const int tid  = threadIdx.x;
    const int lane = tid & 31;
    const int wave = tid >> 5;
    const int rb   = wave >> 1;                 // row block 0..3
    const int nh   = wave & 1;                  // column half 0..1
    const int hi   = (lane >> 4) & 1;
    const int l15  = lane & 15;
    const size_t wgrow0 = (size_t)blockIdx.x * MTILE;

    for (int i = tid; i < OUT_DIM * LEAF; i += 256) wleaf[i] = Wl[i];
    sleaf[tid] = 0.0f;

    v8f acc[8];
#pragma unroll
    for (int t = 0; t < 8; ++t)
#pragma unroll
        for (int r = 0; r < 8; ++r) acc[t][r] = 0.0f;

    const unsigned wldsbase = (unsigned)(size_t)(void*)wbuf;

    // ---- staging helpers ----
    auto stageW = [&](int kt, int buf) {
        const char* wsrc = (const char*)Wswz + (size_t)kt * 16384;
#if USE_ASYNC_COPY
#pragma unroll
        for (int i = 0; i < 4; ++i) {
            const unsigned goff   = (unsigned)(tid + i * 256) * 16u;
            const unsigned ldsoff = wldsbase + (unsigned)buf * 16384u + goff;
            asm volatile("global_load_async_to_lds_b128 %0, %1, %2"
                         :: "v"(ldsoff), "v"(goff), "s"(wsrc)
                         : "memory");
        }
#else
        uint4* wdst = (uint4*)(wbuf + (size_t)buf * 8192);
#pragma unroll
        for (int i = 0; i < 4; ++i) {
            const int u4 = tid + i * 256;
            wdst[u4] = ((const uint4*)wsrc)[u4];
        }
#endif
    };
    // X chunk (64 rows x 32 K, +bias) -> A fragments (cvt bf16).
    // A 16x32 bf16, this code's k order (B same): lane(l<16)=row, reg r: K=2r,2r+1; lanes>=16: +16.
    auto stageX = [&](int kt, int buf) {
        const int kbase = kt * 32;
        __bf16* adst = abuf + (size_t)buf * 4096;
#pragma unroll
        for (int i = 0; i < 4; ++i) {
            const int u   = tid + i * 256;
            const int arb = u >> 8;
            const int l   = (u >> 3) & 31;
            const int r   = u & 7;
            const size_t row = wgrow0 + (size_t)(arb * 16 + (l & 15));
            const int k0  = kbase + 2 * r + ((l & 16) ? 16 : 0);
            const float f0 = (k0 == 0) ? 1.0f
                           : ((k0     < KDIM) ? X[row * IN_DIM + (k0 - 1)] : 0.0f);
            const float f1 =  ((k0 + 1 < KDIM) ? X[row * IN_DIM + k0]       : 0.0f);
            bf16x2 pk; pk.x = (__bf16)f0; pk.y = (__bf16)f1;
            *(bf16x2*)(adst + (size_t)u * 2) = pk;
            if (kt + 1 < NCHUNK)
                __builtin_prefetch(&X[row * IN_DIM + kbase + 32], 0, 3);
        }
    };

    // ---- double-buffered K pipeline ----
    stageW(0, 0);
    stageX(0, 0);
    for (int kt = 0; kt < NCHUNK; ++kt) {
        const int cur = kt & 1;
#if USE_ASYNC_COPY
        asm volatile("s_wait_asynccnt 0" ::: "memory");  // own cur-chunk asyncs done
#endif
        __syncthreads();   // staging(cur) visible; compute(kt-1) done in all waves
        if (kt + 1 < NCHUNK) {
            stageW(kt + 1, cur ^ 1);   // DMA overlaps the WMMAs below
            stageX(kt + 1, cur ^ 1);
        }

        const v16bf a = *(const v16bf*)(abuf + (size_t)cur * 4096
                                             + (size_t)(rb * 32 + lane) * 16);
        const __bf16* wb = wbuf + (size_t)cur * 8192;
#pragma unroll
        for (int tt = 0; tt < 8; ++tt) {
            const int t = nh * 8 + tt;
            const v16bf b = *(const v16bf*)(wb + (size_t)(t * 32 + lane) * 16);
            acc[tt] = __builtin_amdgcn_wmma_f32_16x16x32_bf16(
                false, a, false, b, (short)0, acc[tt], false, false);
        }
    }

    // sigmoid -> LDS p matrix. C tile: reg r, lanes<16: (row r, col lane);
    // lanes>=16: (row r+8, col lane-16).
    float* pw = p_lds + (size_t)rb * 16 * NCOLS;
#pragma unroll
    for (int tt = 0; tt < 8; ++tt)
#pragma unroll
        for (int r = 0; r < 8; ++r) {
            const float v = acc[tt][r];
            pw[(r + (hi ? 8 : 0)) * NCOLS + ((nh * 8 + tt) * 16 + l15)] =
                1.0f / (1.0f + __expf(-v));
        }
    __syncthreads();

    // ---- tree walk: lane owns leaves L = lane*8 + j; wave owns 8 rows ----
    float sacc[8];
#pragma unroll
    for (int j = 0; j < 8; ++j) sacc[j] = 0.0f;

    for (int rr = 0; rr < OUT_ROWS_PER_WAVE; ++rr) {
        const int row = wave * OUT_ROWS_PER_WAVE + rr;      // 0..63 within WG
        const float* pr = p_lds + (size_t)row * NCOLS;
        float mu = 1.0f, p;
        p = pr[0];                 mu *= ((lane >> 4) & 1) ? (1.0f - p) : p;
        p = pr[1  + (lane >> 4)];  mu *= ((lane >> 3) & 1) ? (1.0f - p) : p;
        p = pr[3  + (lane >> 3)];  mu *= ((lane >> 2) & 1) ? (1.0f - p) : p;
        p = pr[7  + (lane >> 2)];  mu *= ((lane >> 1) & 1) ? (1.0f - p) : p;
        p = pr[15 + (lane >> 1)];  mu *= ( lane       & 1) ? (1.0f - p) : p;
        const float p5 = pr[31 + lane];
        float mu6[2] = { mu * p5, mu * (1.0f - p5) };
        float mu7[4], leafmu[8];
#pragma unroll
        for (int b = 0; b < 2; ++b) {
            const float p6 = pr[63 + 2 * lane + b];
            mu7[2 * b]     = mu6[b] * p6;
            mu7[2 * b + 1] = mu6[b] * (1.0f - p6);
        }
#pragma unroll
        for (int bc = 0; bc < 4; ++bc) {
            const float p7 = pr[127 + 4 * lane + bc];
            leafmu[2 * bc]     = mu7[bc] * p7;
            leafmu[2 * bc + 1] = mu7[bc] * (1.0f - p7);
        }
        float y[OUT_DIM];
#pragma unroll
        for (int o = 0; o < OUT_DIM; ++o) y[o] = 0.0f;
#pragma unroll
        for (int j = 0; j < 8; ++j) {
            const float m = leafmu[j];
            sacc[j] += m;
            const int L = lane * 8 + j;
#pragma unroll
            for (int o = 0; o < OUT_DIM; ++o) y[o] += m * wleaf[o * LEAF + L];
        }
#pragma unroll
        for (int off = 16; off > 0; off >>= 1)
#pragma unroll
            for (int o = 0; o < OUT_DIM; ++o) y[o] += __shfl_down(y[o], off, 32);
        if (lane == 0) {
            const size_t gr = wgrow0 + (size_t)row;
#pragma unroll
            for (int o = 0; o < OUT_DIM; ++o) out[gr * OUT_DIM + o] = y[o];
        }
    }

    // fold per-lane leaf sums -> workgroup LDS -> one global atomic per leaf
#pragma unroll
    for (int j = 0; j < 8; ++j) atomicAdd(&sleaf[lane * 8 + j], sacc[j]);
    __syncthreads();
    atomicAdd(&Sleaf_g[tid], sleaf[tid]);
}

__global__ void sdt_finalize(const float* __restrict__ Sleaf_g,
                             float* __restrict__ out) {
    __shared__ float S[511];           // heap: node i children 2i+1, 2i+2; leaves 255..510
    if (threadIdx.x == 0) {
        for (int i = 0; i < LEAF; ++i) S[255 + i] = Sleaf_g[i];
        for (int i = 254; i >= 0; --i) S[i] = S[2 * i + 1] + S[2 * i + 2];
        float pen = 0.0f, coeff = LAMDA;
        int base = 0;
        for (int l = 0; l < 8; ++l) {
            const int n = 1 << l;
            for (int m = 0; m < n; ++m) {
                const int i = base + m;
                const float inv = 1.0f / S[i];
                const float aL = S[2 * i + 1] * inv;
                const float aR = S[2 * i + 2] * inv;
                pen -= 0.5f * coeff * (__logf(aL) + __logf(1.0f - aL) +
                                       __logf(aR) + __logf(1.0f - aR));
            }
            base += n;
            coeff *= 0.5f;
        }
        out[(size_t)B_TOTAL * OUT_DIM] = pen;
    }
}

extern "C" void kernel_launch(void* const* d_in, const int* in_sizes, int n_in,
                              void* d_out, int out_size, void* d_ws, size_t ws_size,
                              hipStream_t stream) {
    (void)in_sizes; (void)n_in; (void)out_size; (void)ws_size;
    const float* X  = (const float*)d_in[0];
    const float* Wi = (const float*)d_in[1];
    const float* Wl = (const float*)d_in[2];
    float* out   = (float*)d_out;
    float*  Sleaf = (float*)d_ws;
    bf16x2* Wswz  = (bf16x2*)((char*)d_ws + WS_WSWZ_OFF);

    sdt_zero<<<1, 256, 0, stream>>>(Sleaf);
    sdt_prep_w<<<dim3((WUNITS + 255) / 256), 256, 0, stream>>>(Wi, Wswz);
    sdt_main<<<dim3(B_TOTAL / MTILE), 256, SM_BYTES, stream>>>(X, Wswz, Wl, out, Sleaf);
    sdt_finalize<<<1, 32, 0, stream>>>(Sleaf, out);
}
